// MoELayer_52003464020210
// MI455X (gfx1250) — matrix operand order repaired and checked by reading, not compiled, not run on hardware
//
#include <hip/hip_runtime.h>

#define HDIM 2048   // hidden size
#define MR   512    // routed intermediate
#define MSH  1024   // shared intermediate (M * n_shared)
#define NE   8      // routed experts
#define NT   2048   // tokens (B*S)
#define TOPK 2

typedef __attribute__((ext_vector_type(16))) _Float16 v16h;
typedef __attribute__((ext_vector_type(8)))  _Float16 h8;
typedef __attribute__((ext_vector_type(8)))  float    v8f;

static __device__ __forceinline__ v8f wmma16(v16h a, v16h b, v8f c) {
  // D = A(16x32 f16) * B(32x16 f16) + C(f32)
  return __builtin_amdgcn_wmma_f32_16x16x32_f16(false, a, false, b, (short)0, c, false, false);
}

static __device__ __forceinline__ v16h mkfrag(h8 lo, h8 hi) {
  v16h r;
#pragma unroll
  for (int j = 0; j < 8; ++j) { r[j] = lo[j]; r[j + 8] = hi[j]; }
  return r;
}

// fast sigmoid: v_exp_f32 + v_add + v_rcp_f32 (no IEEE divide expansion)
static __device__ __forceinline__ float fast_sigmoid(float v) {
  return __builtin_amdgcn_rcpf(1.f + __expf(-v));
}

// ---------------- routing: logits + top-2 + softmax + expert counts ----------------
__global__ __launch_bounds__(256) void route_kernel(
    const float* __restrict__ x, const float* __restrict__ gw,
    int* __restrict__ tIdx, float* __restrict__ tW, int* __restrict__ cnt)
{
  const int lane = threadIdx.x & 31;
  const int t = blockIdx.x * 8 + (threadIdx.x >> 5);
  if (t >= NT) return;
  float acc[NE];
#pragma unroll
  for (int e = 0; e < NE; ++e) acc[e] = 0.f;
  const float* xr = x + (size_t)t * HDIM;
  for (int h = lane; h < HDIM; h += 32) {
    const float xv = xr[h];
#pragma unroll
    for (int e = 0; e < NE; ++e) acc[e] += xv * gw[e * HDIM + h];
  }
#pragma unroll
  for (int e = 0; e < NE; ++e) {
    float v = acc[e];
#pragma unroll
    for (int s = 16; s >= 1; s >>= 1) v += __shfl_xor(v, s, 32);
    acc[e] = v;
  }
  if (lane == 0) {
    int i0 = 0; float v0 = acc[0];
#pragma unroll
    for (int e = 1; e < NE; ++e) if (acc[e] > v0) { v0 = acc[e]; i0 = e; }
    int i1 = -1; float v1 = -3.0e38f;
#pragma unroll
    for (int e = 0; e < NE; ++e) if (e != i0 && acc[e] > v1) { v1 = acc[e]; i1 = e; }
    const float w0 = fast_sigmoid(v0 - v1);          // softmax over {v0,v1}, v0 = max
    tIdx[2 * t] = i0; tIdx[2 * t + 1] = i1;
    tW[2 * t] = w0;   tW[2 * t + 1] = 1.f - w0;
    atomicAdd(&cnt[i0], 1);
    atomicAdd(&cnt[i1], 1);
  }
}

__global__ void zero_cnt(int* __restrict__ cnt) { if (threadIdx.x < NE) cnt[threadIdx.x] = 0; }

__global__ void scan_kernel(const int* __restrict__ cnt, int* __restrict__ basep,
                            int* __restrict__ cursor) {
  if (threadIdx.x == 0) {
    int s = 0;
    for (int e = 0; e < NE; ++e) { basep[e] = s; s += cnt[e]; }
  }
  if (threadIdx.x < NE) cursor[threadIdx.x] = 0;
}

__global__ __launch_bounds__(256) void scatter_kernel(
    const int* __restrict__ tIdx, const float* __restrict__ tW,
    const int* __restrict__ basep, int* __restrict__ cursor,
    int* __restrict__ perm, float* __restrict__ permw)
{
  const int t = blockIdx.x * 256 + threadIdx.x;
  if (t >= NT) return;
#pragma unroll
  for (int k = 0; k < TOPK; ++k) {
    const int e = tIdx[2 * t + k];
    const int p = atomicAdd(&cursor[e], 1);
    perm[basep[e] + p]  = t;
    permw[basep[e] + p] = tW[2 * t + k];
  }
}

// ------- pack fp32 KxNC weights into f16 WMMA B-fragment layout --------
// packed idx((ct,kt,lane,j)) = ((ct*(K/32)+kt)*32 + lane)*16 + j
// lane 0-15: col = ct*16+lane, K = kt*32 + j ; lane 16-31: K += 16
template<int K, int NC>
__global__ __launch_bounds__(256) void pack_w(const float* __restrict__ Win,
                                              _Float16* __restrict__ Pout) {
  const size_t mat = blockIdx.y;
  const float* W = Win + mat * (size_t)K * NC;
  _Float16*    P = Pout + mat * (size_t)K * NC;
  const int idx = blockIdx.x * 256 + threadIdx.x;
  if (idx >= (K * NC) / 16) return;
  const int lane = idx & 31;
  const int kt   = (idx >> 5) % (K / 32);
  const int ct   = idx / ((K / 32) * 32);
  const int col  = ct * 16 + (lane & 15);
  const int kb   = kt * 32 + ((lane >> 4) << 4);
  h8 lo, hi;
#pragma unroll
  for (int j = 0; j < 8; ++j) lo[j] = (_Float16)W[(size_t)(kb + j) * NC + col];
#pragma unroll
  for (int j = 0; j < 8; ++j) hi[j] = (_Float16)W[(size_t)(kb + 8 + j) * NC + col];
  h8* dst = (h8*)(P + (size_t)idx * 16);
  dst[0] = lo; dst[1] = hi;
}

// ---------------- fused SwiGLU MLP over one 16-token tile ----------------
template<int MD, bool GATHER, bool ATOMIC>
__global__ __launch_bounds__(256) void mlp_kernel(
    const float* __restrict__ x,
    const _Float16* __restrict__ Pg, const _Float16* __restrict__ Pu,
    const _Float16* __restrict__ Pd,
    float* __restrict__ out,
    const int* __restrict__ perm, const float* __restrict__ permw,
    const int* __restrict__ cnt, const int* __restrict__ basep)
{
  constexpr int XS = HDIM + 8;   // padded LDS strides (halves) to avoid bank conflicts
  constexpr int HS = MD + 8;
  __shared__ _Float16 sX[16 * XS];
  __shared__ _Float16 sH[16 * HS];
  __shared__ int   sRow[16];
  __shared__ float sWt[16];

  const int tid  = threadIdx.x;
  const int lane = tid & 31;
  const int wave = tid >> 5;
  const int tile = blockIdx.x;

  int e = 0;
  if (GATHER) {
    e = blockIdx.y;
    const int c = cnt[e];
    if (tile * 16 >= c) return;           // uniform per block
    if (tid < 16) {
      const int base   = basep[e];
      const int nvalid = c - tile * 16;
      const int slot   = base + tile * 16 + ((tid < nvalid) ? tid : 0);
      sRow[tid] = perm[slot];
      sWt[tid]  = (tid < nvalid) ? permw[slot] : 0.f;   // padded lanes contribute 0
    }
  } else {
    if (tid < 16) { sRow[tid] = tile * 16 + tid; sWt[tid] = 1.f; }
  }
  __syncthreads();

  // stage X tile (16 x HDIM) fp32 -> f16 LDS
  for (int i = tid; i < 16 * (HDIM / 8); i += 256) {
    const int row = i / (HDIM / 8);
    const int c8  = (i % (HDIM / 8)) * 8;
    const float* src = x + (size_t)sRow[row] * HDIM + c8;
    h8 v;
#pragma unroll
    for (int j = 0; j < 8; ++j) v[j] = (_Float16)src[j];
    *(h8*)&sX[row * XS + c8] = v;
  }
  __syncthreads();

  const size_t guOff = GATHER ? (size_t)e * HDIM * MD : 0;
  const int off  = (lane >> 4) * 16;    // A-frag: hi half-wave takes K+16
  const int rowA = lane & 15;

  // phase 1: h = silu(x*Wg) * (x*Wu), 64 columns per wave per subchunk
  for (int sc = 0; sc < MD / 512; ++sc) {
    const int colbase = wave * (MD / 8) + sc * 64;
    v8f accG[4], accU[4];
#pragma unroll
    for (int c = 0; c < 4; ++c) { accG[c] = 0.f; accU[c] = 0.f; }
    for (int k0 = 0; k0 < HDIM; k0 += 32) {
      const h8* ap = (const h8*)&sX[rowA * XS + k0 + off];
      const v16h a = mkfrag(ap[0], ap[1]);
      const size_t kidx = (size_t)(k0 >> 5) * 32 + lane;
#pragma unroll
      for (int c = 0; c < 4; ++c) {
        const int ct = (colbase >> 4) + c;
        const size_t bi = guOff + ((size_t)ct * (HDIM / 32) * 32 + kidx) * 16;
        const h8* bg = (const h8*)(Pg + bi);
        accG[c] = wmma16(a, mkfrag(bg[0], bg[1]), accG[c]);
        const h8* bu = (const h8*)(Pu + bi);
        accU[c] = wmma16(a, mkfrag(bu[0], bu[1]), accU[c]);
      }
    }
    const int colL = lane & 15, rowB = (lane >> 4) * 8;   // C layout
#pragma unroll
    for (int c = 0; c < 4; ++c) {
      const int col = colbase + c * 16 + colL;
#pragma unroll
      for (int r = 0; r < 8; ++r) {
        const float g  = accG[c][r];
        const float u  = accU[c][r];
        const float hv = g * fast_sigmoid(g) * u;   // silu(g)*u, rcp-based
        sH[(rowB + r) * HS + col] = (_Float16)hv;
      }
    }
  }
  __syncthreads();

  // phase 2: y = h * Wd, 256 output cols per wave in 2 chunks of 128
  const size_t dOff = GATHER ? (size_t)e * MD * HDIM : 0;
  for (int chunk = 0; chunk < 2; ++chunk) {
    const int colbase = wave * 256 + chunk * 128;
    v8f acc[8];
#pragma unroll
    for (int c = 0; c < 8; ++c) acc[c] = 0.f;
    for (int k0 = 0; k0 < MD; k0 += 32) {
      const h8* ap = (const h8*)&sH[rowA * HS + k0 + off];
      const v16h a = mkfrag(ap[0], ap[1]);
      const size_t kidx = (size_t)(k0 >> 5) * 32 + lane;
#pragma unroll
      for (int c = 0; c < 8; ++c) {
        const int ct = (colbase >> 4) + c;
        const size_t bi = dOff + ((size_t)ct * (MD / 32) * 32 + kidx) * 16;
        const h8* bd = (const h8*)(Pd + bi);
        acc[c] = wmma16(a, mkfrag(bd[0], bd[1]), acc[c]);
      }
    }
    const int colL = lane & 15, rowB = (lane >> 4) * 8;
#pragma unroll
    for (int c = 0; c < 8; ++c) {
      const int col = colbase + c * 16 + colL;
#pragma unroll
      for (int r = 0; r < 8; ++r) {
        const int row = rowB + r;
        const float v = acc[c][r];
        if constexpr (ATOMIC) {
          unsafeAtomicAdd(&out[(size_t)sRow[row] * HDIM + col], v * sWt[row]);
        } else {
          out[(size_t)sRow[row] * HDIM + col] = v;
        }
      }
    }
  }
}

extern "C" void kernel_launch(void* const* d_in, const int* in_sizes, int n_in,
                              void* d_out, int out_size, void* d_ws, size_t ws_size,
                              hipStream_t stream) {
  const float* x   = (const float*)d_in[0];
  const float* gw  = (const float*)d_in[1];
  const float* Wg  = (const float*)d_in[2];
  const float* Wu  = (const float*)d_in[3];
  const float* Wd  = (const float*)d_in[4];
  const float* sWg = (const float*)d_in[5];
  const float* sWu = (const float*)d_in[6];
  const float* sWd = (const float*)d_in[7];
  float* out = (float*)d_out;

  uint8_t* w = (uint8_t*)d_ws;
  int*      cnt    = (int*)w;       w += 256;
  int*      basep  = (int*)w;       w += 256;
  int*      cursor = (int*)w;       w += 256;
  int*      tIdx   = (int*)w;       w += (size_t)NT * 2 * 4;
  float*    tW     = (float*)w;     w += (size_t)NT * 2 * 4;
  int*      perm   = (int*)w;       w += (size_t)NT * 2 * 4;
  float*    permw  = (float*)w;     w += (size_t)NT * 2 * 4;
  _Float16* Pg     = (_Float16*)w;  w += (size_t)NE * HDIM * MR * 2;
  _Float16* Pu     = (_Float16*)w;  w += (size_t)NE * HDIM * MR * 2;
  _Float16* Pd     = (_Float16*)w;  w += (size_t)NE * MR * HDIM * 2;
  _Float16* sPg    = (_Float16*)w;  w += (size_t)HDIM * MSH * 2;
  _Float16* sPu    = (_Float16*)w;  w += (size_t)HDIM * MSH * 2;
  _Float16* sPd    = (_Float16*)w;  w += (size_t)MSH * HDIM * 2;

  zero_cnt<<<1, 32, 0, stream>>>(cnt);

  // weight packing fp32 -> f16 B-fragment layout (all fit in 192MB L2 afterwards)
  pack_w<HDIM, MR ><<<dim3(256, NE), 256, 0, stream>>>(Wg,  Pg);
  pack_w<HDIM, MR ><<<dim3(256, NE), 256, 0, stream>>>(Wu,  Pu);
  pack_w<MR,  HDIM><<<dim3(256, NE), 256, 0, stream>>>(Wd,  Pd);
  pack_w<HDIM, MSH><<<dim3(512, 1),  256, 0, stream>>>(sWg, sPg);
  pack_w<HDIM, MSH><<<dim3(512, 1),  256, 0, stream>>>(sWu, sPu);
  pack_w<MSH, HDIM><<<dim3(512, 1),  256, 0, stream>>>(sWd, sPd);

  route_kernel<<<NT / 8, 256, 0, stream>>>(x, gw, tIdx, tW, cnt);
  scan_kernel<<<1, 32, 0, stream>>>(cnt, basep, cursor);
  scatter_kernel<<<NT / 256, 256, 0, stream>>>(tIdx, tW, basep, cursor, perm, permw);

  // shared expert: dense, plain stores -> fully initializes d_out
  mlp_kernel<MSH, false, false><<<NT / 16, 256, 0, stream>>>(
      x, sPg, sPu, sPd, out, nullptr, nullptr, nullptr, nullptr);
  // routed experts: gathered tokens, weighted atomic combine into d_out
  mlp_kernel<MR, true, true><<<dim3(256, NE), 256, 0, stream>>>(
      x, Pg, Pu, Pd, out, perm, permw, cnt, basep);
}
